// AttentionBlock_8237747274109
// MI455X (gfx1250) — compile-verified, matrix operand order checked by reading, not run on hardware
//
#include <hip/hip_runtime.h>
#include <hip/hip_bf16.h>

typedef __attribute__((ext_vector_type(16))) _Float16 v16h;
typedef __attribute__((ext_vector_type(8)))  _Float16 v8h;
typedef __attribute__((ext_vector_type(8)))  float    v8f;

#define B_   16
#define C_   512
#define N_   1024
#define HEADS 8
#define CH   64
#define QKSCALE 0.35355339059327379f   // 64^-0.25

// ---------------------------------------------------------------------------
// helpers
// ---------------------------------------------------------------------------
__device__ inline v8f wmma16(v16h a, v16h b, v8f c) {
    // D = A(16x32 f16) * B(32x16 f16) + C(16x16 f32)
    return __builtin_amdgcn_wmma_f32_16x16x32_f16(
        /*neg_a=*/false, a, /*neg_b=*/false, b,
        /*c_mod=*/(short)0, c, /*reuse_a=*/false, /*reuse_b=*/false);
}

__device__ inline v8f zero8() {
    v8f z;
#pragma unroll
    for (int i = 0; i < 8; ++i) z[i] = 0.0f;
    return z;
}

// Load a 16x32 f16 fragment (A layout; B layout is the mirror with N in place
// of M) from a [rows][ld] matrix that is contiguous along K.
// lane row index = caller passes (tile_row + (lane&15)); khalf = lane>>4.
__device__ inline v16h frag_ld(const _Float16* base, int row, int ld, int k0, int lane) {
    int kh = (lane >> 4) & 1;
    const _Float16* p = base + (size_t)row * ld + k0 + kh * 8;
    v8h lo = *(const v8h*)p;
    v8h hi = *(const v8h*)(p + 16);
    v16h r;
#pragma unroll
    for (int i = 0; i < 8; ++i) { r[i] = lo[i]; r[i + 8] = hi[i]; }
    return r;
}

__device__ inline float rmax16(float v) {
#pragma unroll
    for (int o = 8; o > 0; o >>= 1) v = fmaxf(v, __shfl_xor(v, o, 32));
    return v;
}
__device__ inline float rsum16(float v) {
#pragma unroll
    for (int o = 8; o > 0; o >>= 1) v += __shfl_xor(v, o, 32);
    return v;
}

// ---------------------------------------------------------------------------
// Kernel 0: convert weights fp32 -> f16
// ---------------------------------------------------------------------------
__global__ void __launch_bounds__(256) k_cvt_weights(
    const float* __restrict__ wq, const float* __restrict__ wp,
    _Float16* __restrict__ wqh, _Float16* __restrict__ wph) {
    int i = blockIdx.x * 256 + threadIdx.x;
    if (i < 3 * C_ * C_) wqh[i] = (_Float16)wq[i];
    if (i < C_ * C_)     wph[i] = (_Float16)wp[i];
}

// ---------------------------------------------------------------------------
// Kernel 1: group norm -> xnT [b][n][c] in f16
// one block per (batch, group): 16 channels x 1024 spatial = 16384 elems
// ---------------------------------------------------------------------------
__global__ void __launch_bounds__(256) k_groupnorm(
    const float* __restrict__ x, const float* __restrict__ gamma,
    const float* __restrict__ beta, _Float16* __restrict__ xnT) {
    int bg = blockIdx.x;
    int b = bg >> 5, g = bg & 31;
    const float* xp = x + ((size_t)b * C_ + g * 16) * N_;
    int tid = threadIdx.x;

    float s = 0.f, s2 = 0.f;
    for (int i = tid; i < 16 * N_; i += 256) {
        float v = xp[i];
        s += v; s2 += v * v;
    }
    __shared__ float rs[256], rs2[256];
    rs[tid] = s; rs2[tid] = s2;
    __syncthreads();
    for (int off = 128; off > 0; off >>= 1) {
        if (tid < off) { rs[tid] += rs[tid + off]; rs2[tid] += rs2[tid + off]; }
        __syncthreads();
    }
    const float inv = 1.0f / (16.0f * N_);
    float mean = rs[0] * inv;
    float var  = rs2[0] * inv - mean * mean;
    float rstd = rsqrtf(var + 1e-5f);

    for (int i = tid; i < 16 * N_; i += 256) {
        int cl = i >> 10;            // channel within group
        int n  = i & (N_ - 1);
        int c  = g * 16 + cl;
        float vv = (xp[i] - mean) * rstd * gamma[c] + beta[c];
        xnT[((size_t)b * N_ + n) * C_ + c] = (_Float16)vv;
    }
}

// ---------------------------------------------------------------------------
// Kernel 2: QKV GEMM.  qkv[o,n] = sum_c Wqkv[o,c] * xnT[n,c]  (+bias, +scale)
// 8 waves / block, each wave computes a 32x32 output patch (2x2 WMMA tiles).
// Block covers 64(M) x 128(N). Grid (1536/64, 1024/128, B).
// Writes q,k -> [head][n][c] (scaled), v -> [head][c][n], all f16.
// ---------------------------------------------------------------------------
__global__ void __launch_bounds__(256) k_qkv_gemm(
    const _Float16* __restrict__ wq, const _Float16* __restrict__ xnT,
    const float* __restrict__ bias,
    _Float16* __restrict__ qb, _Float16* __restrict__ kb, _Float16* __restrict__ vb) {
    int b    = blockIdx.z;
    int wid  = threadIdx.x >> 5;
    int lane = threadIdx.x & 31;
    int wm   = wid & 1, wn = wid >> 1;
    int m0   = blockIdx.x * 64 + wm * 32;
    int n0   = blockIdx.y * 128 + wn * 32;
    const _Float16* xb = xnT + (size_t)b * N_ * C_;
    int rl = lane & 15;

    v8f acc00 = zero8(), acc01 = zero8(), acc10 = zero8(), acc11 = zero8();
    for (int kk = 0; kk < C_; kk += 32) {
        if (kk + 32 < C_) {  // prefetch next K-slice (global_prefetch_b8)
            __builtin_prefetch(wq + (size_t)(m0 + rl) * C_ + kk + 32, 0, 3);
            __builtin_prefetch(xb + (size_t)(n0 + rl) * C_ + kk + 32, 0, 3);
        }
        v16h a0 = frag_ld(wq, m0 + rl,      C_, kk, lane);
        v16h a1 = frag_ld(wq, m0 + 16 + rl, C_, kk, lane);
        v16h b0 = frag_ld(xb, n0 + rl,      C_, kk, lane);
        v16h b1 = frag_ld(xb, n0 + 16 + rl, C_, kk, lane);
        acc00 = wmma16(a0, b0, acc00);
        acc01 = wmma16(a0, b1, acc01);
        acc10 = wmma16(a1, b0, acc10);
        acc11 = wmma16(a1, b1, acc11);
    }

#pragma unroll
    for (int i = 0; i < 2; ++i) {
#pragma unroll
        for (int j = 0; j < 2; ++j) {
            v8f acc = (i == 0) ? (j == 0 ? acc00 : acc01) : (j == 0 ? acc10 : acc11);
            int obase = m0 + i * 16;
            int sect  = obase / C_;              // 0=q 1=k 2=v
            int head  = (obase % C_) / CH;       // constant within tile
            int hg    = b * HEADS + head;
            int n     = n0 + j * 16 + (lane & 15);
#pragma unroll
            for (int r = 0; r < 8; ++r) {
                int o = obase + ((lane >= 16) ? 8 : 0) + r;
                int c = o % CH;
                float val = acc[r] + bias[o];
                if (sect == 0)
                    qb[((size_t)hg * N_ + n) * CH + c] = (_Float16)(val * QKSCALE);
                else if (sect == 1)
                    kb[((size_t)hg * N_ + n) * CH + c] = (_Float16)(val * QKSCALE);
                else
                    vb[((size_t)hg * CH + c) * N_ + n] = (_Float16)val;
            }
        }
    }
}

// ---------------------------------------------------------------------------
// Kernel 3: attention, single-pass online softmax (flash-style).
// One wave owns 16 query rows. Per 32-wide s-step:
//   S = Q K^T (4 WMMA), m_new = max(m, rowmax(S)), f = exp(m-m_new),
//   O *= f, l = l*f + rowsum(P), O += P V (4 WMMA).
// P converted D-frag -> A-frag through a per-wave LDS slot.
// Grid (128 heads, 1024/128). Output hT [b][t][c] f16.
// ---------------------------------------------------------------------------
__global__ void __launch_bounds__(256) k_attention(
    const _Float16* __restrict__ qb, const _Float16* __restrict__ kb,
    const _Float16* __restrict__ vb, _Float16* __restrict__ hT) {
    int hg   = blockIdx.x;            // global head
    int b    = hg >> 3, hh = hg & 7;
    int wid  = threadIdx.x >> 5;
    int lane = threadIdx.x & 31;
    int t0   = blockIdx.y * 128 + wid * 16;
    int rl   = lane & 15;

    const _Float16* q = qb + (size_t)hg * N_ * CH;
    const _Float16* k = kb + (size_t)hg * N_ * CH;
    const _Float16* v = vb + (size_t)hg * CH * N_;

    __shared__ _Float16 plds[8][16 * 32];
    _Float16* myp = plds[wid];

    // Q fragments for the wave's 16 rows (K = c, two 32-chunks)
    v16h aq0 = frag_ld(q, t0 + rl, CH, 0,  lane);
    v16h aq1 = frag_ld(q, t0 + rl, CH, 32, lane);

    float m[8], lsum[8];
#pragma unroll
    for (int r = 0; r < 8; ++r) { m[r] = -1e30f; lsum[r] = 0.f; }

    v8f O0 = zero8(), O1 = zero8(), O2 = zero8(), O3 = zero8();

    for (int s0 = 0; s0 < N_; s0 += 32) {
        if (s0 + 32 < N_) {  // prefetch next K/V rows (global_prefetch_b8)
            __builtin_prefetch(k + (size_t)(s0 + 32 + rl) * CH, 0, 3);
            __builtin_prefetch(v + (size_t)rl * N_ + s0 + 32, 0, 3);
        }
        // ---- S = Q K^T for 32 key columns ----------------------------------
        v8f S0 = zero8(), S1 = zero8();
        v16h bk;
        bk = frag_ld(k, s0 + rl,      CH, 0,  lane); S0 = wmma16(aq0, bk, S0);
        bk = frag_ld(k, s0 + rl,      CH, 32, lane); S0 = wmma16(aq1, bk, S0);
        bk = frag_ld(k, s0 + 16 + rl, CH, 0,  lane); S1 = wmma16(aq0, bk, S1);
        bk = frag_ld(k, s0 + 16 + rl, CH, 32, lane); S1 = wmma16(aq1, bk, S1);

        // ---- online softmax update ----------------------------------------
        float fcor[8];
#pragma unroll
        for (int r = 0; r < 8; ++r) {
            float mx   = rmax16(fmaxf(S0[r], S1[r]));
            float mnew = fmaxf(m[r], mx);
            float f    = __expf(m[r] - mnew);
            m[r] = mnew;
            float p0 = __expf(S0[r] - mnew);
            float p1 = __expf(S1[r] - mnew);
            lsum[r] = lsum[r] * f + rsum16(p0 + p1);
            fcor[r] = f;
            int row = (lane < 16) ? r : r + 8;
            int col = lane & 15;
            myp[row * 32 + col]      = (_Float16)p0;
            myp[row * 32 + 16 + col] = (_Float16)p1;
        }
        // rescale O accumulators (row mapping identical to S)
#pragma unroll
        for (int r = 0; r < 8; ++r) {
            O0[r] *= fcor[r]; O1[r] *= fcor[r];
            O2[r] *= fcor[r]; O3[r] *= fcor[r];
        }

        // ---- O += P * V ----------------------------------------------------
        v16h ap = frag_ld(myp, rl, 32, 0, lane);  // P as A-frag (K = s, 32 wide)
        v16h bv;
        bv = frag_ld(v,  0 + rl, N_, s0, lane); O0 = wmma16(ap, bv, O0);
        bv = frag_ld(v, 16 + rl, N_, s0, lane); O1 = wmma16(ap, bv, O1);
        bv = frag_ld(v, 32 + rl, N_, s0, lane); O2 = wmma16(ap, bv, O2);
        bv = frag_ld(v, 48 + rl, N_, s0, lane); O3 = wmma16(ap, bv, O3);
    }

    // ---- normalize and store hT[b][t][head*64+c] -------------------------
    _Float16* hout = hT + (size_t)b * N_ * C_ + hh * CH;
#pragma unroll
    for (int ct = 0; ct < 4; ++ct) {
        v8f O = (ct == 0) ? O0 : (ct == 1) ? O1 : (ct == 2) ? O2 : O3;
        int c = ct * 16 + (lane & 15);
#pragma unroll
        for (int r = 0; r < 8; ++r) {
            int row = (lane < 16) ? r : r + 8;
            float val = O[r] / lsum[r];
            hout[(size_t)(t0 + row) * C_ + c] = (_Float16)val;
        }
    }
}

// ---------------------------------------------------------------------------
// Kernel 4: proj GEMM + bias + residual.  out[b,o,n] = x + Wp*h + bp
// Same tiling as k_qkv_gemm; M=512. Grid (512/64, 1024/128, B).
// ---------------------------------------------------------------------------
__global__ void __launch_bounds__(256) k_proj_gemm(
    const _Float16* __restrict__ wp, const _Float16* __restrict__ hT,
    const float* __restrict__ bias, const float* __restrict__ x,
    float* __restrict__ out) {
    int b    = blockIdx.z;
    int wid  = threadIdx.x >> 5;
    int lane = threadIdx.x & 31;
    int wm   = wid & 1, wn = wid >> 1;
    int m0   = blockIdx.x * 64 + wm * 32;
    int n0   = blockIdx.y * 128 + wn * 32;
    const _Float16* hb = hT + (size_t)b * N_ * C_;
    int rl = lane & 15;

    v8f acc00 = zero8(), acc01 = zero8(), acc10 = zero8(), acc11 = zero8();
    for (int kk = 0; kk < C_; kk += 32) {
        if (kk + 32 < C_) {
            __builtin_prefetch(wp + (size_t)(m0 + rl) * C_ + kk + 32, 0, 3);
            __builtin_prefetch(hb + (size_t)(n0 + rl) * C_ + kk + 32, 0, 3);
        }
        v16h a0 = frag_ld(wp, m0 + rl,      C_, kk, lane);
        v16h a1 = frag_ld(wp, m0 + 16 + rl, C_, kk, lane);
        v16h b0 = frag_ld(hb, n0 + rl,      C_, kk, lane);
        v16h b1 = frag_ld(hb, n0 + 16 + rl, C_, kk, lane);
        acc00 = wmma16(a0, b0, acc00);
        acc01 = wmma16(a0, b1, acc01);
        acc10 = wmma16(a1, b0, acc10);
        acc11 = wmma16(a1, b1, acc11);
    }

#pragma unroll
    for (int i = 0; i < 2; ++i) {
#pragma unroll
        for (int j = 0; j < 2; ++j) {
            v8f acc = (i == 0) ? (j == 0 ? acc00 : acc01) : (j == 0 ? acc10 : acc11);
            int obase = m0 + i * 16;
            int n = n0 + j * 16 + (lane & 15);
#pragma unroll
            for (int r = 0; r < 8; ++r) {
                int o = obase + ((lane >= 16) ? 8 : 0) + r;
                size_t idx = ((size_t)b * C_ + o) * N_ + n;
                out[idx] = x[idx] + acc[r] + bias[o];
            }
        }
    }
}

// ---------------------------------------------------------------------------
// launch
// ---------------------------------------------------------------------------
extern "C" void kernel_launch(void* const* d_in, const int* in_sizes, int n_in,
                              void* d_out, int out_size, void* d_ws, size_t ws_size,
                              hipStream_t stream) {
    const float* x     = (const float*)d_in[0];
    const float* gamma = (const float*)d_in[1];
    const float* beta  = (const float*)d_in[2];
    const float* w_qkv = (const float*)d_in[3];
    const float* b_qkv = (const float*)d_in[4];
    const float* w_prj = (const float*)d_in[5];
    const float* b_prj = (const float*)d_in[6];
    float* out = (float*)d_out;

    const size_t BNC = (size_t)B_ * N_ * C_;          // 8,388,608 elems
    _Float16* xnT = (_Float16*)d_ws;
    _Float16* qb  = xnT + BNC;
    _Float16* kb  = qb + BNC;
    _Float16* vb  = kb + BNC;
    _Float16* hTb = vb + BNC;
    _Float16* wqh = hTb + BNC;
    _Float16* wph = wqh + (size_t)3 * C_ * C_;

    k_cvt_weights<<<(3 * C_ * C_ + 255) / 256, 256, 0, stream>>>(w_qkv, w_prj, wqh, wph);
    k_groupnorm<<<B_ * 32, 256, 0, stream>>>(x, gamma, beta, xnT);
    k_qkv_gemm<<<dim3((3 * C_) / 64, N_ / 128, B_), 256, 0, stream>>>(
        wqh, xnT, b_qkv, qb, kb, vb);
    k_attention<<<dim3(B_ * HEADS, N_ / 128), 256, 0, stream>>>(qb, kb, vb, hTb);
    k_proj_gemm<<<dim3(C_ / 64, N_ / 128, B_), 256, 0, stream>>>(
        wph, hTb, b_prj, x, out);
}
